// NetVLAD_75557064672007
// MI455X (gfx1250) — matrix operand order repaired and checked by reading, not compiled, not run on hardware
//
#include <hip/hip_runtime.h>
#include <cstdint>
#include <cstddef>

typedef __bf16 bf16_t;
typedef __attribute__((ext_vector_type(16))) __bf16 v16bf;
typedef __attribute__((ext_vector_type(8)))  __bf16 v8bf;
typedef __attribute__((ext_vector_type(8)))  float  v8f;
typedef __attribute__((ext_vector_type(4)))  int    v4i;

#define NB   64
#define NPIX 1024      // H*W
#define ND   512
#define NK   64
#define EPSF 1e-12f

// One-instruction XOR butterfly within wave32: DS_SWIZZLE_B32 group-of-32,
// and_mask=0x1f, or_mask=0, xor_mask=m  (no bpermute address setup).
#define SWZ_XOR_F(v, m)                                                        \
  __builtin_bit_cast(float, __builtin_amdgcn_ds_swizzle(                       \
                                __builtin_bit_cast(int, (v)), 0x1f | ((m) << 10)))

// CDNA5 async global->LDS staging path (ASYNCcnt-tracked).
#if defined(__has_builtin)
#if __has_builtin(__builtin_amdgcn_global_load_async_to_lds_b128) && \
    __has_builtin(__builtin_amdgcn_s_wait_asynccnt)
#define USE_ASYNC_LDS 1
#endif
#endif
#ifndef USE_ASYNC_LDS
#define USE_ASYNC_LDS 0
#endif

#if USE_ASYNC_LDS
typedef __attribute__((address_space(1))) v4i* gas1_v4i;   // global <4 x i32>*
typedef __attribute__((address_space(3))) v4i* as3_v4i;    // LDS    <4 x i32>*
// Build AS-qualified pointers from generic HIP pointers via integer casts.
__device__ __forceinline__ gas1_v4i to_global_v4i(const void* p) {
  return (gas1_v4i)(uintptr_t)p;
}
__device__ __forceinline__ as3_v4i to_lds_v4i(const void* p) {
  // generic shared address: LDS byte offset lives in the low 32 bits
  return (as3_v4i)(unsigned)(uintptr_t)p;
}
#endif

// ---------------------------------------------------------------------------
// k_prep: w[D,K] f32 -> wT[K,D] bf16 ; zero asum[B*K], gss[B]
// ---------------------------------------------------------------------------
__global__ __launch_bounds__(256) void k_prep(const float* __restrict__ w,
                                              bf16_t* __restrict__ wT,
                                              float* __restrict__ asum,
                                              float* __restrict__ gss) {
  const int idx = blockIdx.x * 256 + threadIdx.x;   // exactly 32768 threads
  const int k = idx >> 9;
  const int d = idx & 511;
  wT[idx] = (bf16_t)w[d * NK + k];
  if (idx < NB * NK) asum[idx] = 0.f;
  if (idx < NB)      gss[idx]  = 0.f;
}

// ---------------------------------------------------------------------------
// k_assign: s = x * w (WMMA bf16), softmax over K=64, store aT[b][k][n] bf16,
//           accumulate asum[b][k]. One wave per 16-pixel tile.
// ---------------------------------------------------------------------------
__global__ __launch_bounds__(256) void k_assign(const float* __restrict__ x,
                                                const bf16_t* __restrict__ wT,
                                                bf16_t* __restrict__ aT,
                                                float* __restrict__ asum) {
  const int lane = threadIdx.x & 31;
  const int wv   = threadIdx.x >> 5;
  const int half = lane >> 4;     // lane half selects K sub-range per ISA layout
  const int l16  = lane & 15;

  const int t  = blockIdx.x * 8 + wv;   // 16-pixel tile id, 0..4095
  const int b  = t >> 6;                // 64 tiles per batch
  const int p0 = (t & 63) << 4;

  const float* xrow = x + (size_t)(b * NPIX + p0 + l16) * ND;

  v8f c[4] = {};
  for (int d0 = 0; d0 < ND; d0 += 32) {
    // A fragment (16x32 bf16): elems 0..15 = K d0+16h.. (contiguous)
    v16bf a;
    const int ka = d0 + (half << 4);
#pragma unroll
    for (int i = 0; i < 8; ++i) {
      a[i]     = (bf16_t)xrow[ka + i];
      a[8 + i] = (bf16_t)xrow[ka + 8 + i];
    }
#pragma unroll
    for (int kt = 0; kt < 4; ++kt) {
      // B fragment (32x16 bf16): lane = column (cluster), 16 contiguous K (=d)
      const v16bf bfrag =
          *(const v16bf*)(wT + (size_t)(kt * 16 + l16) * ND + d0 + (half << 4));
      c[kt] = __builtin_amdgcn_wmma_f32_16x16x32_bf16(
          false, a, false, bfrag, (short)0, c[kt], false, false);
    }
  }

  // softmax per pixel row: row = (r, lane-half); clusters = 4 kt-tiles x 16 lanes
#pragma unroll
  for (int r = 0; r < 8; ++r) {
    float m = fmaxf(fmaxf(c[0][r], c[1][r]), fmaxf(c[2][r], c[3][r]));
    m = fmaxf(m, SWZ_XOR_F(m, 1));
    m = fmaxf(m, SWZ_XOR_F(m, 2));
    m = fmaxf(m, SWZ_XOR_F(m, 4));
    m = fmaxf(m, SWZ_XOR_F(m, 8));
    const float e0 = __expf(c[0][r] - m);
    const float e1 = __expf(c[1][r] - m);
    const float e2 = __expf(c[2][r] - m);
    const float e3 = __expf(c[3][r] - m);
    float sm = e0 + e1 + e2 + e3;
    sm += SWZ_XOR_F(sm, 1);
    sm += SWZ_XOR_F(sm, 2);
    sm += SWZ_XOR_F(sm, 4);
    sm += SWZ_XOR_F(sm, 8);
    const float inv = 1.0f / sm;
    c[0][r] = e0 * inv; c[1][r] = e1 * inv; c[2][r] = e2 * inv; c[3][r] = e3 * inv;
  }

  // store transposed assignments + column sums
#pragma unroll
  for (int kt = 0; kt < 4; ++kt) {
    const int k = kt * 16 + l16;
    v8bf st;
    float cs = 0.f;
#pragma unroll
    for (int r = 0; r < 8; ++r) { st[r] = (bf16_t)c[kt][r]; cs += c[kt][r]; }
    // pixel = p0 + 8*half + r  -> contiguous 8 bf16
    *(v8bf*)(aT + (size_t)(b * NK + k) * NPIX + p0 + (half << 3)) = st;
    cs += SWZ_XOR_F(cs, 16);
    if (half == 0) atomicAdd(&asum[b * NK + k], cs);
  }
}

// ---------------------------------------------------------------------------
// k_vlad: vkd[b][k][d] = sum_n aT[b][k][n]*x[b][n][d] + asum[b][k]*C[d][k]
// block = 4 waves, one batch x 64 d-columns; x tile transposed to bf16 in LDS.
// ---------------------------------------------------------------------------
__global__ __launch_bounds__(128) void k_vlad(const float* __restrict__ x,
                                              const bf16_t* __restrict__ aT,
                                              const float* __restrict__ asum,
                                              const float* __restrict__ Cmat,
                                              float* __restrict__ vkd) {
  __shared__ alignas(64) bf16_t xT[64][32];   // [d-local][pixel-local] bf16
#if USE_ASYNC_LDS
  __shared__ alignas(16) float xraw[32][64];  // raw fp32 tile staged via ASYNC
#endif

  const int tid  = threadIdx.x;
  const int lane = tid & 31;
  const int wv   = tid >> 5;
  const int half = lane >> 4;
  const int l16  = lane & 15;
  const int b     = blockIdx.y;
  const int dbase = blockIdx.x * 64;

  const int pp  = (tid & 15) * 2;    // pixel pair this thread transposes
  const int seg = (tid >> 4) * 8;    // its 8 d-columns

  v8f c[4] = {};
  for (int p0 = 0; p0 < NPIX; p0 += 32) {
#if USE_ASYNC_LDS
    // stage raw fp32 tile [32p x 64d] with async global->LDS b128 copies
#pragma unroll
    for (int j = 0; j < 4; ++j) {
      const int chunk = tid + j * 128;          // 512 x 16B chunks
      const int p  = chunk >> 4;
      const int dq = (chunk & 15) * 4;
      const float* gp = x + (size_t)(b * NPIX + p0 + p) * ND + dbase + dq;
      __builtin_amdgcn_global_load_async_to_lds_b128(
          to_global_v4i(gp), to_lds_v4i(&xraw[p][dq]), 0, 0);
    }
    __builtin_amdgcn_s_wait_asynccnt(0);
    __syncthreads();
    const float* r0 = &xraw[pp][seg];
    const float* r1 = &xraw[pp + 1][seg];
#else
    const float* r0 = x + (size_t)(b * NPIX + p0 + pp) * ND + dbase + seg;
    const float* r1 = r0 + ND;
    if (p0 + 32 < NPIX) __builtin_prefetch(r0 + (size_t)32 * ND, 0, 0);
#endif
    // transpose-convert: pack two pixels' bf16 into one 32-bit LDS store
#pragma unroll
    for (int i = 0; i < 8; ++i) {
      const unsigned lo = __builtin_bit_cast(unsigned short, (bf16_t)r0[i]);
      const unsigned hi = __builtin_bit_cast(unsigned short, (bf16_t)r1[i]);
      *(unsigned*)&xT[seg + i][pp] = lo | (hi << 16);
    }
    __syncthreads();

    // B fragment: column d = dbase + wv*16 + l16, 16 contiguous pixels per half
    const v16bf bfrag = *(const v16bf*)(&xT[wv * 16 + l16][half << 4]);
#pragma unroll
    for (int mt = 0; mt < 4; ++mt) {
      const bf16_t* ap =
          aT + (size_t)(b * NK + mt * 16 + l16) * NPIX + p0 + (half << 4);
      const v8bf lo = *(const v8bf*)ap;
      const v8bf hi = *(const v8bf*)(ap + 8);
      v16bf a;
#pragma unroll
      for (int i = 0; i < 8; ++i) { a[i] = lo[i]; a[8 + i] = hi[i]; }
      c[mt] = __builtin_amdgcn_wmma_f32_16x16x32_bf16(
          false, a, false, bfrag, (short)0, c[mt], false, false);
    }
    __syncthreads();
  }

  const int d = dbase + wv * 16 + l16;
#pragma unroll
  for (int mt = 0; mt < 4; ++mt) {
#pragma unroll
    for (int r = 0; r < 8; ++r) {
      const int k = mt * 16 + r + 8 * half;
      const float v = c[mt][r] + asum[b * NK + k] * Cmat[(size_t)d * NK + k];
      vkd[(size_t)(b * NK + k) * ND + d] = v;
    }
  }
}

// ---------------------------------------------------------------------------
// k_rownorm: ss[b,k] = sum_d vkd^2 ; gss[b] += ss/(ss+eps)  (post-intra-norm ssq)
// ---------------------------------------------------------------------------
__global__ __launch_bounds__(128) void k_rownorm(const float* __restrict__ vkd,
                                                 float* __restrict__ ss,
                                                 float* __restrict__ gss) {
  __shared__ float red[4];
  const int row = blockIdx.x;                 // b*64 + k
  const float* vp = vkd + (size_t)row * ND;
  float s = 0.f;
  for (int i = threadIdx.x; i < ND; i += 128) { const float v = vp[i]; s += v * v; }
  s += SWZ_XOR_F(s, 1);
  s += SWZ_XOR_F(s, 2);
  s += SWZ_XOR_F(s, 4);
  s += SWZ_XOR_F(s, 8);
  s += SWZ_XOR_F(s, 16);
  if ((threadIdx.x & 31) == 0) red[threadIdx.x >> 5] = s;
  __syncthreads();
  if (threadIdx.x == 0) {
    const float tot = red[0] + red[1] + red[2] + red[3];
    ss[row] = tot;
    atomicAdd(&gss[row >> 6], tot / (tot + EPSF));
  }
}

// ---------------------------------------------------------------------------
// k_final: out[b, d*K + k] = vkd[b,k,d] * rsqrt(ss+eps) * rsqrt(gss+eps)
// ---------------------------------------------------------------------------
__global__ __launch_bounds__(128) void k_final(const float* __restrict__ vkd,
                                               const float* __restrict__ ss,
                                               const float* __restrict__ gss,
                                               float* __restrict__ out) {
  const int row = blockIdx.x;
  const int b = row >> 6, k = row & 63;
  const float inv = rsqrtf(ss[row] + EPSF) * rsqrtf(gss[b] + EPSF);
  const float* vp = vkd + (size_t)row * ND;
  float* op = out + (size_t)b * (ND * NK) + k;
  for (int d = threadIdx.x; d < ND; d += 128) op[(size_t)d * NK] = vp[d] * inv;
}

// ---------------------------------------------------------------------------
extern "C" void kernel_launch(void* const* d_in, const int* in_sizes, int n_in,
                              void* d_out, int out_size, void* d_ws, size_t ws_size,
                              hipStream_t stream) {
  const float* x    = (const float*)d_in[0];   // [B,H,W,D] f32
  const float* w    = (const float*)d_in[1];   // [D,K] f32
  const float* Cmat = (const float*)d_in[2];   // [D,K] f32
  float* out = (float*)d_out;

  uint8_t* ws = (uint8_t*)d_ws;
  bf16_t* aT   = (bf16_t*)ws;                               // B*K*N bf16  (8 MB)
  float*  vkd  = (float*)(ws + (size_t)8 * 1024 * 1024);    // B*K*D f32   (8 MB)
  float*  ss   = vkd + (size_t)NB * NK * ND;                // B*K f32
  float*  gss  = ss + NB * NK;                              // B f32
  float*  asum = gss + NB;                                  // B*K f32
  bf16_t* wT   = (bf16_t*)(asum + NB * NK);                 // K*D bf16 (32B-aligned)

  k_prep   <<<128, 256, 0, stream>>>(w, wT, asum, gss);
  k_assign <<<512, 256, 0, stream>>>(x, wT, aT, asum);
  k_vlad   <<<dim3(8, 64), 128, 0, stream>>>(x, aT, asum, Cmat, vkd);
  k_rownorm<<<NB * NK, 128, 0, stream>>>(vkd, ss, gss);
  k_final  <<<NB * NK, 128, 0, stream>>>(vkd, ss, gss, out);

  (void)in_sizes; (void)n_in; (void)out_size; (void)ws_size;
}
